// residualBlock_72919954752305
// MI455X (gfx1250) — compile-verified, matrix operand order checked by reading, not run on hardware
//
#include <hip/hip_runtime.h>
#include <hip/hip_bf16.h>

typedef __attribute__((ext_vector_type(16))) _Float16 v16h;
typedef __attribute__((ext_vector_type(8)))  float    v8f;
typedef __attribute__((ext_vector_type(4)))  unsigned int u32x4;

#define NPTS 200000
#define HCH  64
#define CCH  128
#define KEEPK 63
#define TILES (NPTS / 16)   // 12500 (even)

__device__ __forceinline__ float eluf(float v) {
  return v > 0.0f ? v : (__expf(v) - 1.0f);
}

// 32B contiguous fragment load (16 f16). Works for global or LDS pointers;
// through a __shared__-derived pointer this lowers to ds_load_b128 pairs.
__device__ __forceinline__ v16h load_b_frag(const _Float16* p) {
  v16h b;
  ((u32x4*)&b)[0] = *(const u32x4*)(p);
  ((u32x4*)&b)[1] = *(const u32x4*)(p + 8);
  return b;
}

// A-matrix per-lane load: halves [0..7] at p, halves [16..23] at p+16
// (ISA 16-bit A 16x32 layout: lane<16 K={0..7,16..23}, lane>=16 K offset +8).
__device__ __forceinline__ v16h load_a_pair(const _Float16* __restrict__ p) {
  v16h a;
  ((u32x4*)&a)[0] = *(const u32x4*)(p);
  ((u32x4*)&a)[1] = *(const u32x4*)(p + 16);
  return a;
}

__device__ __forceinline__ v8f wmma16(v16h a, v16h b, v8f c) {
  return __builtin_amdgcn_wmma_f32_16x16x32_f16(
      /*neg_a=*/false, a, /*neg_b=*/false, b,
      /*c_mod=*/(short)0, c, /*reuse_a=*/false, /*reuse_b=*/false);
}

// 32-bit LDS byte offset of a generic pointer into a __shared__ array.
__device__ __forceinline__ unsigned lds_addr_of(const _Float16* p) {
  return (unsigned)(unsigned long long)
      (__attribute__((address_space(3))) const _Float16*)p;
}

// Async-DMA one ko's worth of Wb fragments (4096 halves = 8 KB) into LDS.
// 512 x 16B chunks, 256 threads x 2 chunks, tracked by ASYNCcnt.
__device__ __forceinline__ void stage_wb(const _Float16* __restrict__ wbFrag,
                                         int ko, _Float16* ldsbuf, int t) {
  const _Float16* g = wbFrag + ko * 4096;
#pragma unroll
  for (int c = 0; c < 2; ++c) {
    int chunk = t + c * 256;
    unsigned lds_a = lds_addr_of(ldsbuf + chunk * 8);
    const _Float16* ga = g + chunk * 8;
    asm volatile("global_load_async_to_lds_b128 %0, %1, off"
                 :: "v"(lds_a), "v"(ga) : "memory");
  }
}

// ---------------------------------------------------------------------------
// Weight swizzle: f32 [rows x srcN] -> f16 fragments, 512 halves per fragment.
// fragid = kokc*numNt + nt; source row = kokc*32 + k; col = nt*16 + n.
// dst[fragid*512 + lane*16 + e], inverse of the WMMA B operand layout.
// ---------------------------------------------------------------------------
__global__ __launch_bounds__(256) void swizzleW_kernel(
    const float* __restrict__ src, _Float16* __restrict__ dst,
    int numNt, int srcN, int totalHalves) {
  int tid = blockIdx.x * blockDim.x + threadIdx.x;
  if (tid >= totalHalves) return;
  int fragid = tid >> 9;
  int within = tid & 511;
  int lane = within >> 4;
  int e    = within & 15;
  int n  = lane & 15;
  int hi = lane >> 4;
  int j    = e >> 1;
  int jl   = j & 3;
  int jh   = j >> 2;
  int half = e & 1;
  int k = 16 * jh + 8 * hi + 2 * jl + half;   // K within 32-chunk
  int kokc = fragid / numNt;
  int nt   = fragid % numNt;
  int row = kokc * 32 + k;
  int col = nt * 16 + n;
  dst[tid] = (_Float16)src[row * srcN + col];
}

// ---------------------------------------------------------------------------
// Phase 1: g16 = f16( elu( elu(x + y) @ Wa ) )   [N,64]
// ---------------------------------------------------------------------------
__global__ __launch_bounds__(256) void phase1_kernel(
    const float* __restrict__ x, const float* __restrict__ y,
    const _Float16* __restrict__ waFrag, _Float16* __restrict__ g16) {
  int tile = blockIdx.x * (blockDim.x >> 5) + (threadIdx.x >> 5);
  if (tile >= TILES) return;
  int lane  = threadIdx.x & 31;
  int m     = lane & 15;
  int hi    = lane >> 4;
  int kbase = hi * 8;
  int arow  = tile * 16 + m;

  v8f acc[4] = {};
#pragma unroll
  for (int kc = 0; kc < 4; ++kc) {
    const float* xp = x + arow * CCH + kc * 32 + kbase;
    const float* yp = y + arow * CCH + kc * 32 + kbase;
    v16h a;
#pragma unroll
    for (int e = 0; e < 16; ++e) {
      int off = (e < 8) ? e : (e + 8);            // K = kbase + {0..7,16..23}
      a[e] = (_Float16)eluf(xp[off] + yp[off]);
    }
#pragma unroll
    for (int nt = 0; nt < 4; ++nt) {
      v16h b = load_b_frag(waFrag + (kc * 4 + nt) * 512 + lane * 16);
      acc[nt] = wmma16(a, b, acc[nt]);
    }
  }
#pragma unroll
  for (int nt = 0; nt < 4; ++nt) {
#pragma unroll
    for (int j = 0; j < 8; ++j) {
      int r = tile * 16 + j + 8 * hi;
      g16[r * HCH + nt * 16 + m] = (_Float16)eluf(acc[nt][j]);
    }
  }
}

// ---------------------------------------------------------------------------
// Phase 2: acc[i,:] = sum_k gather(g16, nidx[k,i]) @ Wb[k]   (63 offsets)
// Each wave computes TWO 16-row tiles (32 rows) so every staged B fragment
// feeds two WMMAs. Wb fragments are double-buffered in LDS via
// global_load_async_to_lds_b128; the next ko's DMA overlaps this ko's WMMAs.
// g16 (25.6 MB f16) stays L2-resident across all 63 gather passes.
// ---------------------------------------------------------------------------
__global__ __launch_bounds__(256) void phase2_kernel(
    const _Float16* __restrict__ g16, const _Float16* __restrict__ wbFrag,
    const int* __restrict__ nidx, float* __restrict__ accOut) {
  __shared__ _Float16 lds_wb[2][4096];   // 2 x 8 KB double buffer

  int t      = threadIdx.x;
  int waveid = t >> 5;
  int lane   = t & 31;
  int w      = blockIdx.x * 8 + waveid;  // global wave id
  int tile0  = w * 2;
  bool active = (tile0 < TILES);         // TILES is even -> tile0+1 also valid

  int m     = lane & 15;
  int hi    = lane >> 4;
  int kbase = hi * 8;
  int base  = tile0 * 16;                // 32 consecutive rows

  v8f acc0[4] = {};
  v8f acc1[4] = {};

  stage_wb(wbFrag, 0, &lds_wb[0][0], t);

  int buf = 0;
  for (int ko = 0; ko < KEEPK; ++ko) {
    asm volatile("s_wait_asynccnt 0x0" ::: "memory");
    __syncthreads();                       // staged buffer visible to all waves
    if (ko + 1 < KEEPK) stage_wb(wbFrag, ko + 1, &lds_wb[buf ^ 1][0], t);
    const _Float16* wb = &lds_wb[buf][0];

    if (active) {
      int idx0 = nidx[ko * NPTS + base + m];
      int idx1 = nidx[ko * NPTS + base + 16 + m];
#pragma unroll
      for (int kc = 0; kc < 2; ++kc) {
        v16h a0 = {}, a1 = {};
        if (idx0 >= 0) a0 = load_a_pair(g16 + idx0 * HCH + kc * 32 + kbase);
        if (idx1 >= 0) a1 = load_a_pair(g16 + idx1 * HCH + kc * 32 + kbase);
#pragma unroll
        for (int nt = 0; nt < 4; ++nt) {
          v16h b = load_b_frag(wb + (kc * 4 + nt) * 512 + lane * 16);
          acc0[nt] = wmma16(a0, b, acc0[nt]);
          acc1[nt] = wmma16(a1, b, acc1[nt]);
        }
      }
    }
    buf ^= 1;
  }

  if (active) {
#pragma unroll
    for (int nt = 0; nt < 4; ++nt) {
#pragma unroll
      for (int j = 0; j < 8; ++j) {
        int r0 = base + j + 8 * hi;
        int r1 = base + 16 + j + 8 * hi;
        accOut[r0 * HCH + nt * 16 + m] = acc0[nt][j];
        accOut[r1 * HCH + nt * 16 + m] = acc1[nt][j];
      }
    }
  }
}

// ---------------------------------------------------------------------------
// Phase 3: out = x + elu(acc) @ Wc   [N,128]
// ---------------------------------------------------------------------------
__global__ __launch_bounds__(256) void phase3_kernel(
    const float* __restrict__ accIn, const _Float16* __restrict__ wcFrag,
    const float* __restrict__ x, float* __restrict__ out) {
  int tile = blockIdx.x * (blockDim.x >> 5) + (threadIdx.x >> 5);
  if (tile >= TILES) return;
  int lane  = threadIdx.x & 31;
  int m     = lane & 15;
  int hi    = lane >> 4;
  int kbase = hi * 8;
  int arow  = tile * 16 + m;

  v8f acc[8] = {};
#pragma unroll
  for (int kc = 0; kc < 2; ++kc) {
    const float* p = accIn + arow * HCH + kc * 32 + kbase;
    v16h a;
#pragma unroll
    for (int e = 0; e < 16; ++e) {
      int off = (e < 8) ? e : (e + 8);
      a[e] = (_Float16)eluf(p[off]);
    }
#pragma unroll
    for (int nt = 0; nt < 8; ++nt) {
      v16h b = load_b_frag(wcFrag + (kc * 8 + nt) * 512 + lane * 16);
      acc[nt] = wmma16(a, b, acc[nt]);
    }
  }
#pragma unroll
  for (int nt = 0; nt < 8; ++nt) {
#pragma unroll
    for (int j = 0; j < 8; ++j) {
      int r = tile * 16 + j + 8 * hi;
      int c = nt * 16 + m;
      out[r * CCH + c] = x[r * CCH + c] + acc[nt][j];
    }
  }
}

// ---------------------------------------------------------------------------
extern "C" void kernel_launch(void* const* d_in, const int* in_sizes, int n_in,
                              void* d_out, int out_size, void* d_ws, size_t ws_size,
                              hipStream_t stream) {
  (void)in_sizes; (void)n_in; (void)out_size; (void)ws_size;
  const float* x    = (const float*)d_in[0];   // [N,128]
  const float* y    = (const float*)d_in[1];   // [N,128]
  const float* Wa   = (const float*)d_in[2];   // [128,64]
  const float* Wb   = (const float*)d_in[3];   // [125,64,64] (first 63 used)
  const float* Wc   = (const float*)d_in[4];   // [64,128]
  const int*   nidx = (const int*)d_in[5];     // [63,N]
  float* out = (float*)d_out;

  // Workspace layout (all offsets 256B-aligned):
  //   g16   : N*64 f16   = 25,600,000 B
  //   acc   : N*64 f32   = 51,200,000 B
  //   waFrag: 16  frags  =     16,384 B
  //   wbFrag: 504 frags  =    516,096 B
  //   wcFrag: 16  frags  =     16,384 B
  char* ws = (char*)d_ws;
  _Float16* g16    = (_Float16*)(ws);
  float*    accBuf = (float*)  (ws + 25600000);
  _Float16* waFrag = (_Float16*)(ws + 76800000);
  _Float16* wbFrag = (_Float16*)(ws + 76816384);
  _Float16* wcFrag = (_Float16*)(ws + 77332480);

  // Weight swizzle into WMMA B-operand fragment layout (f32 -> f16).
  swizzleW_kernel<<<(16 * 512 + 255) / 256, 256, 0, stream>>>(
      Wa, waFrag, /*numNt=*/4, /*srcN=*/64, 16 * 512);
  swizzleW_kernel<<<(504 * 512 + 255) / 256, 256, 0, stream>>>(
      Wb, wbFrag, /*numNt=*/4, /*srcN=*/64, 504 * 512);
  swizzleW_kernel<<<(16 * 512 + 255) / 256, 256, 0, stream>>>(
      Wc, wcFrag, /*numNt=*/8, /*srcN=*/128, 16 * 512);

  int blocks1 = (TILES + 7) / 8;           // 1 tile per wave
  int blocks2 = (TILES / 2 + 7) / 8;       // 2 tiles per wave
  phase1_kernel<<<blocks1, 256, 0, stream>>>(x, y, waFrag, g16);
  phase2_kernel<<<blocks2, 256, 0, stream>>>(g16, wbFrag, nidx, accBuf);
  phase3_kernel<<<blocks1, 256, 0, stream>>>(accBuf, wcFrag, x, out);
}
